// MultiHeadSelfAttention_34119220199718
// MI455X (gfx1250) — matrix operand order repaired and checked by reading, not compiled, not run on hardware
//
#include <hip/hip_runtime.h>
#include <hip/hip_bf16.h>
#include <stdint.h>

// Problem constants (from reference): B=2, T=2048, D=1024, H=16, Dh=64
#define B_   2
#define T_   2048
#define D_   1024
#define H_   16
#define DH_  64
#define LOG2E 1.4426950408889634f

typedef __bf16 bf16;
typedef __attribute__((ext_vector_type(16))) __bf16 v16bf;
typedef __attribute__((ext_vector_type(8)))  float  v8f;
typedef __attribute__((ext_vector_type(4))) unsigned int u32x4;
typedef __attribute__((ext_vector_type(8))) int         i32x8;
typedef __attribute__((ext_vector_type(4))) int         i32x4;

// fp32 -> bf16 round-to-nearest-even (pure bit ops; no bf16 arithmetic needed)
__device__ __forceinline__ bf16 f2bf(float f) {
  uint32_t u = __builtin_bit_cast(uint32_t, f);
  u += 0x7FFFu + ((u >> 16) & 1u);
  uint16_t h = (uint16_t)(u >> 16);
  return __builtin_bit_cast(bf16, h);
}

// Low 32 bits of a generic pointer to LDS == LDS byte offset (flat aperture
// layout: addr[31:0] is the LDS address).
__device__ __forceinline__ uint32_t lds_addr32(const void* p) {
  return (uint32_t)(uintptr_t)p;
}

// Per-lane async global->LDS copy of 16 bytes (ASYNCcnt tracked).
__device__ __forceinline__ void async_copy_b128(uint32_t lds_off, const void* g) {
  unsigned long long ga = (unsigned long long)(uintptr_t)g;
  asm volatile("global_load_async_to_lds_b128 %0, %1, off"
               :: "v"(lds_off), "v"(ga) : "memory");
}
__device__ __forceinline__ void wait_asynccnt0() {
  asm volatile("s_wait_asynccnt 0x0" ::: "memory");
}

// WMMA 16x16x32 bf16 fragment index maps used here (wave32; k order is this code's choice):
// A (16x32, MxK): lane m = L%16; element j -> k = j + 16*(L/16)
// B (32x16, KxN): lane n = L%16; element j -> k = j + 16*(L/16)
// C/D (16x16):    lane n = L%16; reg r    -> m = r + 8*(L/16)
#define AKIDX(j, hl) ((j) + ((hl) << 4))
#define BKIDX(j, hl) ((j) + ((hl) << 4))

__device__ __forceinline__ v8f wmma_bf16(v16bf a, v16bf b, v8f c) {
  return __builtin_amdgcn_wmma_f32_16x16x32_bf16(false, a, false, b, (short)0, c,
                                                 false, false);
}

// ---------------------------------------------------------------------------
// Kernel 1: QKV projection. C[m,e] = sum_d x[m,d] * w_qkv[e,d]
// M=4096 (B*T), K=1024, N=3072. Output scattered to q/k/v ws (B,H,T,Dh) bf16.
// Q is pre-scaled by Dh^-0.5 = 0.125 (exact in bf16).
// (A/W staging stays through VGPRs: fp32->bf16 conversion happens in regs.)
// ---------------------------------------------------------------------------
__global__ __launch_bounds__(128)
void qkv_gemm_kernel(const float* __restrict__ x, const float* __restrict__ w_qkv,
                     bf16* __restrict__ q_ws, bf16* __restrict__ k_ws,
                     bf16* __restrict__ v_ws) {
  __shared__ __align__(16) bf16 sA[64][40];  // 64 rows x 32 k (+8 pad)
  __shared__ __align__(16) bf16 sW[64][40];
  const int tid  = threadIdx.x;
  const int wave = tid >> 5, lane = tid & 31;
  const int hl   = lane >> 4, l16 = lane & 15;
  const int m0 = blockIdx.y * 64, n0 = blockIdx.x * 64;

  v8f acc[4];
#pragma unroll
  for (int i = 0; i < 4; ++i) acc[i] = v8f{0.f,0.f,0.f,0.f,0.f,0.f,0.f,0.f};

  for (int k0 = 0; k0 < D_; k0 += 32) {
    __syncthreads();
#pragma unroll
    for (int i = 0; i < 4; ++i) {
      int cid = tid + 128 * i;              // 512 chunks of 4 floats
      int row = cid >> 3, col = (cid & 7) * 4;
      float4 a = *(const float4*)(x + (size_t)(m0 + row) * D_ + k0 + col);
      sA[row][col + 0] = f2bf(a.x); sA[row][col + 1] = f2bf(a.y);
      sA[row][col + 2] = f2bf(a.z); sA[row][col + 3] = f2bf(a.w);
      float4 w = *(const float4*)(w_qkv + (size_t)(n0 + row) * D_ + k0 + col);
      sW[row][col + 0] = f2bf(w.x); sW[row][col + 1] = f2bf(w.y);
      sW[row][col + 2] = f2bf(w.z); sW[row][col + 3] = f2bf(w.w);
    }
    __syncthreads();

    v16bf af;
#pragma unroll
    for (int j = 0; j < 16; ++j) af[j] = sA[wave * 16 + l16][AKIDX(j, hl)];
#pragma unroll
    for (int ns = 0; ns < 4; ++ns) {
      v16bf bfm;
#pragma unroll
      for (int j = 0; j < 16; ++j) bfm[j] = sW[ns * 16 + l16][BKIDX(j, hl)];
      acc[ns] = wmma_bf16(af, bfm, acc[ns]);
    }
  }

  // Scatter: e -> (which, h, dh); m -> (b, t)
#pragma unroll
  for (int ns = 0; ns < 4; ++ns) {
    int n = n0 + ns * 16 + l16;
    int which = n >> 10, d = n & 1023, h = d >> 6, dh = d & 63;
    bf16* dst = (which == 0) ? q_ws : (which == 1 ? k_ws : v_ws);
    float sc = (which == 0) ? 0.125f : 1.0f;
#pragma unroll
    for (int r = 0; r < 8; ++r) {
      int m = m0 + wave * 16 + r + hl * 8;
      int b = m >> 11, t = m & (T_ - 1);
      dst[((size_t)(b * H_ + h) * T_ + t) * DH_ + dh] = f2bf(acc[ns][r] * sc);
    }
  }
}

// ---------------------------------------------------------------------------
// Kernel 2: Flash attention. One block = 8 waves = 128 query rows of one
// (b,h). S^T = K*Q^T so softmax state is per-lane; O^T = V^T*P^T so the
// online-softmax correction is a per-lane scalar on the accumulators.
// K tile staged via per-lane GLOBAL_LOAD_ASYNC_TO_LDS_B128 (ASYNCcnt);
// V tile staged via TDM TENSOR_LOAD_TO_LDS (TENSORcnt) with LDS row padding
// reproduced by the descriptor's pad feature.
// ---------------------------------------------------------------------------
__global__ __launch_bounds__(256)
void attn_kernel(const bf16* __restrict__ q_ws, const bf16* __restrict__ k_ws,
                 const bf16* __restrict__ v_ws,
                 const unsigned char* __restrict__ pad_mask,
                 const int* __restrict__ causal_p, bf16* __restrict__ o_ws) {
  __shared__ __align__(16) bf16 sK[32][72];   // 32 keys x 64 dh (+8 pad)
  __shared__ __align__(16) bf16 sV[32][72];
  __shared__ float sPad[32];

  const int tid  = threadIdx.x;
  const int wave = tid >> 5, lane = tid & 31;
  const int hl   = lane >> 4, l16 = lane & 15;
  const int bx = blockIdx.x;
  const int qt = bx & (T_ / 128 - 1);
  const int bh = bx >> 4;                 // T_/128 == 16
  const int h  = bh & (H_ - 1), b = bh >> 4;
  const int q0w  = qt * 128;
  const int q0   = q0w + wave * 16;
  const int qpos = q0 + l16;
  const int causal = causal_p[0];
  const float NEG_INF = -__builtin_inff();

  const size_t headoff = (size_t)(b * H_ + h) * T_ * DH_;
  const bf16* kh = k_ws + headoff;
  const bf16* vh = v_ws + headoff;
  const uint32_t sV_base = lds_addr32(&sV[0][0]);

  // Q fragments (B operand of S^T): element j -> dh = BKIDX(j,hl) (+32)
  v16bf qf0, qf1;
  {
    const bf16* qrow = q_ws + headoff + (size_t)qpos * DH_;
#pragma unroll
    for (int j = 0; j < 16; ++j) {
      qf0[j] = qrow[BKIDX(j, hl)];
      qf1[j] = qrow[32 + BKIDX(j, hl)];
    }
  }

  v8f o[4];
#pragma unroll
  for (int i = 0; i < 4; ++i) o[i] = v8f{0.f,0.f,0.f,0.f,0.f,0.f,0.f,0.f};
  float m_run = NEG_INF, l_run = 0.f;

  const int k_end = causal ? (q0w + 128) : T_;
  for (int kb = 0; kb < k_end; kb += 32) {
    __syncthreads();
    {  // K tile: 256 async 16B lane-copies, bypassing VGPRs
      int row = tid >> 3, c = (tid & 7) * 8;
      async_copy_b128(lds_addr32(&sK[row][c]),
                      kh + (size_t)(kb + row) * DH_ + c);
      if (kb + 32 < k_end) {
        __builtin_prefetch(kh + (size_t)(kb + 32 + row) * DH_ + c, 0, 3);
        __builtin_prefetch(vh + (size_t)(kb + 32 + row) * DH_ + c, 0, 3);
      }
    }
    if (wave == 0) {
      // V tile via Tensor Data Mover: 2D tile 64x32 elems of 2B,
      // row stride 64 elems in memory, +16B LDS pad per 128B row.
      unsigned long long ga = (unsigned long long)(uintptr_t)(vh + (size_t)kb * DH_);
      u32x4 g0 = {1u,                                   // count=1 (valid D#)
                  sV_base,                              // lds_addr
                  (uint32_t)ga,                         // global_addr[31:0]
                  (uint32_t)(ga >> 32) | (2u << 30)};   // ga[56:32] | type=2
      i32x8 g1 = {(1 << 16) | (1 << 20) | (4 << 22) | (3 << 25),
                  // data_size=1(2B) | pad_enable | pad_interval=32dw | pad_amount=4dw
                  (DH_ << 16),                          // tensor_dim0 = 64
                  (T_ << 16),                           // tensor_dim1 = 2048
                  (DH_ << 16),                          // tile_dim0 = 64
                  32,                                   // tile_dim1 = 32
                  DH_,                                  // tensor_dim0_stride = 64
                  0, 0};
      i32x4 gz4 = {0, 0, 0, 0};
      i32x8 gz8 = {0, 0, 0, 0, 0, 0, 0, 0};
      __builtin_amdgcn_tensor_load_to_lds(g0, g1, gz4, gz4, gz8, 0);
      __builtin_amdgcn_s_wait_tensorcnt((short)0);
    }
    if (tid < 32)
      sPad[tid] = pad_mask[(size_t)b * T_ + kb + tid] ? NEG_INF : 0.f;
    wait_asynccnt0();
    __syncthreads();

    if (!causal || kb <= q0 + 15) {
      // S^T tiles: lane q = l16; reg r -> key = kb + 16*ks + 8*hl + r
      float ps[2][8];
#pragma unroll
      for (int ks = 0; ks < 2; ++ks) {
        v16bf a0, a1;
        const bf16* krow = &sK[ks * 16 + l16][0];
#pragma unroll
        for (int j = 0; j < 16; ++j) {
          a0[j] = krow[AKIDX(j, hl)];
          a1[j] = krow[32 + AKIDX(j, hl)];
        }
        v8f s = v8f{0.f,0.f,0.f,0.f,0.f,0.f,0.f,0.f};
        s = wmma_bf16(a0, qf0, s);
        s = wmma_bf16(a1, qf1, s);
#pragma unroll
        for (int r = 0; r < 8; ++r) {
          int kloc = ks * 16 + hl * 8 + r;
          int kpos = kb + kloc;
          float val = s[r] + sPad[kloc];
          if (causal && kpos > qpos) val = NEG_INF;
          ps[ks][r] = val;
        }
      }
      // online softmax (per-lane state; partner lane is L^16)
      float tmax = ps[0][0];
#pragma unroll
      for (int ks = 0; ks < 2; ++ks)
#pragma unroll
        for (int r = 0; r < 8; ++r) tmax = fmaxf(tmax, ps[ks][r]);
      tmax = fmaxf(tmax, __shfl_xor(tmax, 16, 32));
      float m_new = fmaxf(m_run, tmax);
      float corr  = exp2f((m_run - m_new) * LOG2E);
      float lsum = 0.f;
#pragma unroll
      for (int ks = 0; ks < 2; ++ks)
#pragma unroll
        for (int r = 0; r < 8; ++r) {
          float p = exp2f((ps[ks][r] - m_new) * LOG2E);
          ps[ks][r] = p;
          lsum += p;
        }
      lsum += __shfl_xor(lsum, 16, 32);
      l_run = l_run * corr + lsum;
      m_run = m_new;
#pragma unroll
      for (int i = 0; i < 4; ++i)
#pragma unroll
        for (int r = 0; r < 8; ++r) o[i][r] *= corr;

      // Build P^T B-fragment: needs key = j + 16*hl; half the values come
      // from the partner lane (L^16) -> single 8-value exchange.
      float pex[8];
#pragma unroll
      for (int r = 0; r < 8; ++r) {
        float xv = hl ? ps[0][r] : ps[1][r];
        pex[r] = __shfl_xor(xv, 16, 32);
      }
      v16bf pf;
#pragma unroll
      for (int j = 0; j < 16; ++j) {
        float pv = (j < 8) ? (hl ? pex[j] : ps[0][j])
                           : (hl ? ps[1][j - 8] : pex[j - 8]);
        pf[j] = f2bf(pv);
      }
      // O^T += V^T * P^T : A = V^T (m = dh, inner = key)
#pragma unroll
      for (int sub = 0; sub < 4; ++sub) {
        v16bf vf;
#pragma unroll
        for (int j = 0; j < 16; ++j)
          vf[j] = sV[AKIDX(j, hl)][sub * 16 + l16];
        o[sub] = wmma_bf16(vf, pf, o[sub]);
      }
    }
  }

  // Epilogue: O^T acc -> O (B,T,D) bf16. Lane q = l16; reg r -> dh
  // consecutive -> 8 bf16 = 16B contiguous store.
  float inv = (l_run > 0.f) ? (1.0f / l_run) : 0.f;
  const int t = qpos;
#pragma unroll
  for (int sub = 0; sub < 4; ++sub) {
    union { uint16_t us[8]; uint4 v; } u;
#pragma unroll
    for (int r = 0; r < 8; ++r)
      u.us[r] = __builtin_bit_cast(uint16_t, f2bf(o[sub][r] * inv));
    size_t idx = (size_t)(b * T_ + t) * D_ + h * DH_ + sub * 16 + hl * 8;
    *(uint4*)(o_ws + idx) = u.v;
  }
}

// ---------------------------------------------------------------------------
// Kernel 3: output projection. out[m,n] = sum_d O[m,d] * w_proj[n,d]
// M=4096, K=1024, N=1024; A already bf16 -> staged with async-to-LDS copies,
// output fp32 row-major = d_out.
// ---------------------------------------------------------------------------
__global__ __launch_bounds__(128)
void proj_gemm_kernel(const bf16* __restrict__ o_ws,
                      const float* __restrict__ w_proj,
                      float* __restrict__ out) {
  __shared__ __align__(16) bf16 sA[64][40];
  __shared__ __align__(16) bf16 sW[64][40];
  const int tid  = threadIdx.x;
  const int wave = tid >> 5, lane = tid & 31;
  const int hl   = lane >> 4, l16 = lane & 15;
  const int m0 = blockIdx.y * 64, n0 = blockIdx.x * 64;

  v8f acc[4];
#pragma unroll
  for (int i = 0; i < 4; ++i) acc[i] = v8f{0.f,0.f,0.f,0.f,0.f,0.f,0.f,0.f};

  for (int k0 = 0; k0 < D_; k0 += 32) {
    __syncthreads();
#pragma unroll
    for (int i = 0; i < 2; ++i) {      // A: 256 async 16B chunks (bf16 source)
      int cid = tid + 128 * i;
      int row = cid >> 2, c = (cid & 3) * 8;
      async_copy_b128(lds_addr32(&sA[row][c]),
                      o_ws + (size_t)(m0 + row) * D_ + k0 + c);
    }
#pragma unroll
    for (int i = 0; i < 4; ++i) {      // W: 512 chunks of 4 floats (convert)
      int cid = tid + 128 * i;
      int row = cid >> 3, col = (cid & 7) * 4;
      float4 w = *(const float4*)(w_proj + (size_t)(n0 + row) * D_ + k0 + col);
      sW[row][col + 0] = f2bf(w.x); sW[row][col + 1] = f2bf(w.y);
      sW[row][col + 2] = f2bf(w.z); sW[row][col + 3] = f2bf(w.w);
    }
    wait_asynccnt0();
    __syncthreads();

    v16bf af;
#pragma unroll
    for (int j = 0; j < 16; ++j) af[j] = sA[wave * 16 + l16][AKIDX(j, hl)];
#pragma unroll
    for (int ns = 0; ns < 4; ++ns) {
      v16bf bfm;
#pragma unroll
      for (int j = 0; j < 16; ++j) bfm[j] = sW[ns * 16 + l16][BKIDX(j, hl)];
      acc[ns] = wmma_bf16(af, bfm, acc[ns]);
    }
  }

#pragma unroll
  for (int ns = 0; ns < 4; ++ns) {
    int n = n0 + ns * 16 + l16;
#pragma unroll
    for (int r = 0; r < 8; ++r) {
      int m = m0 + wave * 16 + r + hl * 8;
      out[(size_t)m * D_ + n] = acc[ns][r];
    }
  }
}

// ---------------------------------------------------------------------------
extern "C" void kernel_launch(void* const* d_in, const int* in_sizes, int n_in,
                              void* d_out, int out_size, void* d_ws,
                              size_t ws_size, hipStream_t stream) {
  (void)in_sizes; (void)n_in; (void)out_size; (void)ws_size;
  const float* x      = (const float*)d_in[0];
  const float* w_qkv  = (const float*)d_in[1];
  const float* w_proj = (const float*)d_in[2];
  const unsigned char* mask = (const unsigned char*)d_in[3];  // jax bool (B,T)
  const int* causal   = (const int*)d_in[4];
  float* out = (float*)d_out;

  const size_t HEAD_BYTES = (size_t)B_ * H_ * T_ * DH_ * sizeof(bf16);  // 8 MB
  char* ws = (char*)d_ws;
  bf16* q_ws = (bf16*)(ws);
  bf16* k_ws = (bf16*)(ws + HEAD_BYTES);
  bf16* v_ws = (bf16*)(ws + 2 * HEAD_BYTES);
  bf16* o_ws = (bf16*)(ws + 3 * HEAD_BYTES);

  qkv_gemm_kernel<<<dim3(3 * D_ / 64, (B_ * T_) / 64), 128, 0, stream>>>(
      x, w_qkv, q_ws, k_ws, v_ws);
  attn_kernel<<<dim3(B_ * H_ * (T_ / 128)), 256, 0, stream>>>(
      q_ws, k_ws, v_ws, mask, causal, o_ws);
  proj_gemm_kernel<<<dim3(D_ / 64, (B_ * T_) / 64), 128, 0, stream>>>(
      o_ws, w_proj, out);
}
